// NetVLADLayer_80453327389112
// MI455X (gfx1250) — matrix operand order repaired and checked by reading, not compiled, not run on hardware
//
#include <hip/hip_runtime.h>
#include <hip/hip_bf16.h>

#define B_   32
#define D_   512
#define N_   1600
#define K_   64
#define NP_  1664          // N padded to 13*128 so every GEMM tile is full
#define EPSN 1e-12f

typedef __attribute__((ext_vector_type(16))) __bf16 v16bf;
typedef __attribute__((ext_vector_type(8)))  __bf16 v8bf;
typedef __attribute__((ext_vector_type(8)))  float  v8f;

// A-fragment (16x32 bf16): lane m=lane&15 holds K = hi*16..+15
static __device__ inline v16bf load_a_frag(const __bf16* __restrict__ rowp, int hi) {
    v8bf lo = *(const v8bf*)(rowp + hi * 16);
    v8bf hh = *(const v8bf*)(rowp + hi * 16 + 8);
    return __builtin_shufflevector(lo, hh, 0,1,2,3,4,5,6,7,8,9,10,11,12,13,14,15);
}

// ---------------- kernel 0: zero global-norm accumulators ----------------
__global__ void k_zero(float* __restrict__ gsq) {
    if (threadIdx.x < B_) gsq[threadIdx.x] = 0.0f;
}

// ---------------- kernel 1: per-pixel inverse L2 norm over channels ------
__global__ __launch_bounds__(256) void k_norm(const float* __restrict__ x,
                                              float* __restrict__ inv_norm) {
    int idx = blockIdx.x * 256 + threadIdx.x;
    if (idx >= B_ * N_) return;
    int b = idx / N_, n = idx - b * N_;
    const float* xb = x + (size_t)b * D_ * N_ + n;
    float ss = 0.0f;
    for (int d = 0; d < D_; ++d) {
        float v = __builtin_nontemporal_load(&xb[(size_t)d * N_]);   // x not reused: keep L2 for bf16
        ss += v * v;
    }
    inv_norm[idx] = 1.0f / fmaxf(sqrtf(ss), EPSN);
}

// ---------------- kernel 2a: conv weight fp32 -> bf16 --------------------
__global__ __launch_bounds__(256) void k_convw(const float* __restrict__ w,
                                               __bf16* __restrict__ wb) {
    int i = blockIdx.x * 256 + threadIdx.x;
    if (i < K_ * D_) wb[i] = (__bf16)w[i];
}

// --------- kernel 2: normalize+convert x, emit (D,N) and (N,D) bf16 ------
__global__ __launch_bounds__(256) void k_xpose(const float* __restrict__ x,
                                               const float* __restrict__ inv_norm,
                                               __bf16* __restrict__ xn,   // (B,D,NP) n-contig
                                               __bf16* __restrict__ xt) { // (B,NP,D) d-contig
    __shared__ __bf16 tile[64][68];
    const int n0 = blockIdx.x * 64, d0 = blockIdx.y * 64, b = blockIdx.z;
    const int col = threadIdx.x & 63, rg = threadIdx.x >> 6;
    const int n = n0 + col;
    const float inv = (n < N_) ? inv_norm[b * N_ + n] : 0.0f;
    #pragma unroll
    for (int r = 0; r < 16; ++r) {
        int row = rg * 16 + r;
        float v = (n < N_)
            ? __builtin_nontemporal_load(&x[((size_t)b * D_ + d0 + row) * N_ + n]) * inv
            : 0.0f;
        __bf16 bv = (__bf16)v;
        xn[((size_t)b * D_ + d0 + row) * NP_ + n] = bv;
        tile[row][col] = bv;                 // tile[d][n]
    }
    __syncthreads();
    #pragma unroll
    for (int r = 0; r < 16; ++r) {
        int nrow = rg * 16 + r;
        xt[((size_t)b * NP_ + n0 + nrow) * D_ + d0 + col] = tile[col][nrow];
    }
}

// ------- kernel 3: GEMM1 (logits = W x X) + bias + softmax over K --------
__global__ __launch_bounds__(256) void k_gemm1_softmax(const __bf16* __restrict__ wb,
                                                       const float* __restrict__ bias,
                                                       const __bf16* __restrict__ xt,
                                                       __bf16* __restrict__ a_out) {
    __shared__ float sbias[K_];
    __shared__ float slog[128][65];          // [n-col][k-row], padded vs bank conflicts
    const int b = blockIdx.y, n0 = blockIdx.x * 128, t = threadIdx.x;
    const int w = t >> 5, lane = t & 31, lm = lane & 15, hi = lane >> 4;
    const int mt = w & 3, wn = w >> 2;       // wave -> 16 k-rows x 64 n-cols

    if (t < K_) sbias[t] = bias[t];
    __syncthreads();

    v8f acc[4];
    #pragma unroll
    for (int j = 0; j < 4; ++j) acc[j] = (v8f){0.f,0.f,0.f,0.f,0.f,0.f,0.f,0.f};

    const __bf16* xtb  = xt + (size_t)b * NP_ * D_;
    const __bf16* arow = wb + (size_t)(mt * 16 + lm) * D_;
    const __bf16* brow[4];
    #pragma unroll
    for (int j = 0; j < 4; ++j) {
        int n = n0 + wn * 64 + j * 16 + lm;
        brow[j] = xtb + (size_t)n * D_ + hi * 16;
    }

    // ping-pong pipeline (unroll-by-2): every buffer is written directly by a
    // load; no register-copy swap, one k-step of load lookahead.
    v16bf afA = load_a_frag(arow, hi);
    v16bf bfA[4];
    #pragma unroll
    for (int j = 0; j < 4; ++j) bfA[j] = *(const v16bf*)(brow[j]);

    int kk = 0;
    for (; kk < D_ - 64; kk += 64) {
        v16bf afB = load_a_frag(arow + kk + 32, hi);
        v16bf bfB[4];
        #pragma unroll
        for (int j = 0; j < 4; ++j) bfB[j] = *(const v16bf*)(brow[j] + kk + 32);
        #pragma unroll
        for (int j = 0; j < 4; ++j)
            acc[j] = __builtin_amdgcn_wmma_f32_16x16x32_bf16(
                false, afA, false, bfA[j], (short)0, acc[j], false, false);
        afA = load_a_frag(arow + kk + 64, hi);
        #pragma unroll
        for (int j = 0; j < 4; ++j) bfA[j] = *(const v16bf*)(brow[j] + kk + 64);
        #pragma unroll
        for (int j = 0; j < 4; ++j)
            acc[j] = __builtin_amdgcn_wmma_f32_16x16x32_bf16(
                false, afB, false, bfB[j], (short)0, acc[j], false, false);
    }
    {   // epilogue: last pair (kk == D_-64)
        v16bf afB = load_a_frag(arow + kk + 32, hi);
        v16bf bfB[4];
        #pragma unroll
        for (int j = 0; j < 4; ++j) bfB[j] = *(const v16bf*)(brow[j] + kk + 32);
        #pragma unroll
        for (int j = 0; j < 4; ++j)
            acc[j] = __builtin_amdgcn_wmma_f32_16x16x32_bf16(
                false, afA, false, bfA[j], (short)0, acc[j], false, false);
        #pragma unroll
        for (int j = 0; j < 4; ++j)
            acc[j] = __builtin_amdgcn_wmma_f32_16x16x32_bf16(
                false, afB, false, bfB[j], (short)0, acc[j], false, false);
    }

    #pragma unroll
    for (int j = 0; j < 4; ++j) {
        int col = wn * 64 + j * 16 + lm;
        #pragma unroll
        for (int r = 0; r < 8; ++r) {
            int row = mt * 16 + hi * 8 + r;
            slog[col][row] = acc[j][r] + sbias[row];
        }
    }
    __syncthreads();

    if (t < 128) {                           // softmax over the 64 clusters
        float m = -3.4e38f;
        for (int r = 0; r < K_; ++r) m = fmaxf(m, slog[t][r]);
        float s = 0.0f;
        for (int r = 0; r < K_; ++r) { float e = __expf(slog[t][r] - m); slog[t][r] = e; s += e; }
        float is = 1.0f / s;
        for (int r = 0; r < K_; ++r) slog[t][r] *= is;
    }
    __syncthreads();

    __bf16* ab = a_out + (size_t)b * K_ * NP_;
    for (int idx = t; idx < K_ * 128; idx += 256) {
        int k = idx >> 7, c = idx & 127;
        ab[(size_t)k * NP_ + n0 + c] = (__bf16)slog[c][k];
    }
}

// ---------------- kernel 4: a_sum[b][k] = sum_{n<N} a ---------------------
__global__ __launch_bounds__(256) void k_asum(const __bf16* __restrict__ a_in,
                                              float* __restrict__ asum) {
    __shared__ float red[256];
    const int k = blockIdx.x, b = blockIdx.y;
    const __bf16* row = a_in + ((size_t)b * K_ + k) * NP_;
    float s = 0.0f;
    for (int n = threadIdx.x; n < N_; n += 256) s += (float)row[n];
    red[threadIdx.x] = s;
    __syncthreads();
    for (int st = 128; st > 0; st >>= 1) {
        if (threadIdx.x < st) red[threadIdx.x] += red[threadIdx.x + st];
        __syncthreads();
    }
    if (threadIdx.x == 0) asum[b * K_ + k] = red[0];
}

// ------- kernel 5: GEMM2 vlad = a x X^T  -  a_sum * centroids ------------
__global__ __launch_bounds__(256) void k_gemm2_vlad(const __bf16* __restrict__ a_in,
                                                    const __bf16* __restrict__ xn,
                                                    const float* __restrict__ asum,
                                                    const float* __restrict__ cent,
                                                    float* __restrict__ vlad) {
    __shared__ float sas[K_];
    const int b = blockIdx.y, d0 = blockIdx.x * 128, t = threadIdx.x;
    const int w = t >> 5, lane = t & 31, lm = lane & 15, hi = lane >> 4;
    const int mt = w & 3, wn = w >> 2;       // wave -> 16 k-rows x 64 d-cols

    if (t < K_) sas[t] = asum[b * K_ + t];
    __syncthreads();

    v8f acc[4];
    #pragma unroll
    for (int j = 0; j < 4; ++j) acc[j] = (v8f){0.f,0.f,0.f,0.f,0.f,0.f,0.f,0.f};

    const __bf16* xnb  = xn + (size_t)b * D_ * NP_;
    const __bf16* arow = a_in + ((size_t)b * K_ + mt * 16 + lm) * NP_;
    const __bf16* brow[4];
    #pragma unroll
    for (int j = 0; j < 4; ++j) {
        int d = d0 + wn * 64 + j * 16 + lm;
        brow[j] = xnb + (size_t)d * NP_ + hi * 16;
    }

    v16bf afA = load_a_frag(arow, hi);
    v16bf bfA[4];
    #pragma unroll
    for (int j = 0; j < 4; ++j) bfA[j] = *(const v16bf*)(brow[j]);

    int nn = 0;
    for (; nn < NP_ - 64; nn += 64) {        // pad region of xn is zero
        v16bf afB = load_a_frag(arow + nn + 32, hi);
        v16bf bfB[4];
        #pragma unroll
        for (int j = 0; j < 4; ++j) bfB[j] = *(const v16bf*)(brow[j] + nn + 32);
        #pragma unroll
        for (int j = 0; j < 4; ++j)
            acc[j] = __builtin_amdgcn_wmma_f32_16x16x32_bf16(
                false, afA, false, bfA[j], (short)0, acc[j], false, false);
        afA = load_a_frag(arow + nn + 64, hi);
        #pragma unroll
        for (int j = 0; j < 4; ++j) bfA[j] = *(const v16bf*)(brow[j] + nn + 64);
        #pragma unroll
        for (int j = 0; j < 4; ++j)
            acc[j] = __builtin_amdgcn_wmma_f32_16x16x32_bf16(
                false, afB, false, bfB[j], (short)0, acc[j], false, false);
    }
    {   // epilogue: last pair (nn == NP_-64)
        v16bf afB = load_a_frag(arow + nn + 32, hi);
        v16bf bfB[4];
        #pragma unroll
        for (int j = 0; j < 4; ++j) bfB[j] = *(const v16bf*)(brow[j] + nn + 32);
        #pragma unroll
        for (int j = 0; j < 4; ++j)
            acc[j] = __builtin_amdgcn_wmma_f32_16x16x32_bf16(
                false, afA, false, bfA[j], (short)0, acc[j], false, false);
        #pragma unroll
        for (int j = 0; j < 4; ++j)
            acc[j] = __builtin_amdgcn_wmma_f32_16x16x32_bf16(
                false, afB, false, bfB[j], (short)0, acc[j], false, false);
    }

    float* vb = vlad + (size_t)b * K_ * D_;
    #pragma unroll
    for (int j = 0; j < 4; ++j) {
        int d = d0 + wn * 64 + j * 16 + lm;
        #pragma unroll
        for (int r = 0; r < 8; ++r) {
            int k = mt * 16 + hi * 8 + r;
            vb[(size_t)k * D_ + d] = acc[j][r] - sas[k] * cent[(size_t)k * D_ + d];
        }
    }
}

// ---------------- kernel 6: intra-normalization + global sumsq ------------
__global__ __launch_bounds__(128) void k_intra(float* __restrict__ vlad,
                                               float* __restrict__ gsq) {
    __shared__ float red[128];
    __shared__ float sinv;
    const int row = blockIdx.x;              // b*K + k
    const int b = row >> 6;
    float* v = vlad + (size_t)row * D_;
    float s = 0.0f;
    for (int i = threadIdx.x; i < D_; i += 128) { float t = v[i]; s += t * t; }
    red[threadIdx.x] = s;
    __syncthreads();
    for (int st = 64; st > 0; st >>= 1) {
        if (threadIdx.x < st) red[threadIdx.x] += red[threadIdx.x + st];
        __syncthreads();
    }
    if (threadIdx.x == 0) {
        float tot = red[0];
        float inv = 1.0f / fmaxf(sqrtf(tot), EPSN);
        sinv = inv;
        atomicAdd(&gsq[b], tot * inv * inv);
    }
    __syncthreads();
    float inv = sinv;
    for (int i = threadIdx.x; i < D_; i += 128) v[i] *= inv;
}

// ---------------- kernel 7: global L2 scale -> output ---------------------
__global__ __launch_bounds__(256) void k_final(const float* __restrict__ vlad,
                                               const float* __restrict__ gsq,
                                               float* __restrict__ out) {
    int idx = blockIdx.x * 256 + threadIdx.x;
    if (idx >= B_ * K_ * D_) return;
    int b = idx >> 15;                        // K_*D_ == 32768
    float val = vlad[idx] * (1.0f / fmaxf(sqrtf(gsq[b]), EPSN));
    __builtin_nontemporal_store(val, &out[idx]);
}

extern "C" void kernel_launch(void* const* d_in, const int* in_sizes, int n_in,
                              void* d_out, int out_size, void* d_ws, size_t ws_size,
                              hipStream_t stream) {
    const float* x      = (const float*)d_in[0];
    const float* conv_w = (const float*)d_in[1];
    const float* conv_b = (const float*)d_in[2];
    const float* cent   = (const float*)d_in[3];
    float* out = (float*)d_out;

    char* ws = (char*)d_ws;
    size_t off = 0;
    auto take = [&](size_t bytes) { char* p = ws + off; off = (off + bytes + 255) & ~(size_t)255; return p; };

    float*  inv_norm = (float*) take((size_t)B_ * N_ * 4);
    __bf16* wb       = (__bf16*)take((size_t)K_ * D_ * 2);
    __bf16* xn       = (__bf16*)take((size_t)B_ * D_ * NP_ * 2);   // (B,D,NP)
    __bf16* xt       = (__bf16*)take((size_t)B_ * NP_ * D_ * 2);   // (B,NP,D)
    __bf16* a_buf    = (__bf16*)take((size_t)B_ * K_ * NP_ * 2);   // (B,K,NP)
    float*  asum     = (float*) take((size_t)B_ * K_ * 4);
    float*  vlad     = (float*) take((size_t)B_ * K_ * D_ * 4);
    float*  gsq      = (float*) take((size_t)B_ * 4);
    (void)ws_size; (void)in_sizes; (void)n_in; (void)out_size;

    k_zero<<<1, 64, 0, stream>>>(gsq);
    k_norm<<<(B_ * N_ + 255) / 256, 256, 0, stream>>>(x, inv_norm);
    k_convw<<<(K_ * D_ + 255) / 256, 256, 0, stream>>>(conv_w, wb);
    k_xpose<<<dim3(NP_ / 64, D_ / 64, B_), 256, 0, stream>>>(x, inv_norm, xn, xt);
    k_gemm1_softmax<<<dim3(NP_ / 128, B_), 256, 0, stream>>>(wb, conv_b, xt, a_buf);
    k_asum<<<dim3(K_, B_), 256, 0, stream>>>(a_buf, asum);
    k_gemm2_vlad<<<dim3(D_ / 128, B_), 256, 0, stream>>>(a_buf, xn, asum, cent, vlad);
    k_intra<<<B_ * K_, 128, 0, stream>>>(vlad, gsq);
    k_final<<<(B_ * K_ * D_ + 255) / 256, 256, 0, stream>>>(vlad, gsq, out);
}
